// DeepseekMoEBlock_63651415327116
// MI455X (gfx1250) — matrix operand order repaired and checked by reading, not compiled, not run on hardware
//
#include <hip/hip_runtime.h>
#include <hip/hip_bf16.h>

typedef __bf16 v16bf __attribute__((ext_vector_type(16)));
typedef float  v8f   __attribute__((ext_vector_type(8)));

#define KT 32
#define BM 128
#define BN 64
#define LDSROW 48   // bf16 elements per LDS row: 96B, 16B-aligned, bank-staggered

union FragBF { uint4 q[2]; v16bf v; };

// fp32 -> bf16, round-half-up (1 add + high-half extract)
__device__ __forceinline__ unsigned short f2bf(float f) {
  unsigned int u = __float_as_uint(f) + 0x8000u;
  return (unsigned short)(u >> 16);
}

// Pack two fp32 -> packed bf16x2 (low = lo, high = hi).
// Prefer v_cvt_pk_bf16_f32 if the toolchain has it; else 2x v_add + v_perm_b32.
__device__ __forceinline__ unsigned int pack_bf16(float lo, float hi) {
#if __has_builtin(__builtin_amdgcn_cvt_pk_bf16_f32)
  auto r = __builtin_amdgcn_cvt_pk_bf16_f32(lo, hi);
  unsigned int u;
  __builtin_memcpy(&u, &r, 4);
  return u;
#else
  unsigned int a = __float_as_uint(lo) + 0x8000u;   // round-half-up
  unsigned int b = __float_as_uint(hi) + 0x8000u;
  // result bytes: [1:0] = a[3:2], [3:2] = b[3:2]  (b is src0 -> indices 4..7)
  return __builtin_amdgcn_perm(b, a, 0x07060302u);
#endif
}

__device__ __forceinline__ float silu_mul(float g, float u) {
  // g * sigmoid(g) * u with fast v_exp + v_rcp (no IEEE divide chain)
  float s = __builtin_amdgcn_rcpf(1.0f + __expf(-g));
  return g * s * u;
}

// ---------------------------------------------------------------------------
// x (fp32) -> bf16 staging copy
// ---------------------------------------------------------------------------
__global__ void cvt_kernel(const float* __restrict__ x, unsigned short* __restrict__ xb,
                           long long n) {
  long long i = ((long long)blockIdx.x * blockDim.x + threadIdx.x) * 4;
  if (i >= n) return;
  float4 v = *(const float4*)(x + i);
  uint2 o;
  o.x = pack_bf16(v.x, v.y);
  o.y = pack_bf16(v.z, v.w);
  *(uint2*)(xb + i) = o;
}

// ---------------------------------------------------------------------------
// Router: fp32 logits, softmax over E=16, top-4, expert counts. 1 wave/token.
// ---------------------------------------------------------------------------
__global__ __launch_bounds__(256) void gate_kernel(
    const float* __restrict__ x, const float* __restrict__ gw, int H, int T,
    int* __restrict__ top_idx, float* __restrict__ top_w, int* __restrict__ counts) {
  int w = threadIdx.x >> 5, lane = threadIdx.x & 31;
  int t = blockIdx.x * 8 + w;
  if (t >= T) return;
  int e = lane & 15, half = lane >> 4;
  const float* xr = x  + (long long)t * H + half * (H / 2);
  const float* gr = gw + (long long)e * H + half * (H / 2);
  float s = 0.f;
  for (int k = 0; k < H / 2; k += 4) {
    float4 xv = *(const float4*)(xr + k);
    float4 gv = *(const float4*)(gr + k);
    s += xv.x * gv.x + xv.y * gv.y + xv.z * gv.z + xv.w * gv.w;
  }
  s += __shfl_xor(s, 16);                       // both halves now hold logit(e)
  float m = s;
  for (int d = 1; d < 16; d <<= 1) m = fmaxf(m, __shfl_xor(m, d));
  float p = __expf(s - m);
  float sum = p;
  for (int d = 1; d < 16; d <<= 1) sum += __shfl_xor(sum, d);
  p /= sum;
  float pv = p;
  for (int k = 0; k < 4; ++k) {                 // iterative argmax over 16 lanes
    float best = pv; int bi = e;
    for (int d = 1; d < 16; d <<= 1) {
      float ob = __shfl_xor(best, d);
      int   oi = __shfl_xor(bi, d);
      if (ob > best || (ob == best && oi < bi)) { best = ob; bi = oi; }
    }
    if (lane == 0) {
      top_idx[t * 4 + k] = bi;
      top_w [t * 4 + k] = best;
      atomicAdd(&counts[bi], 1);
    }
    if (e == bi) pv = -1.0f;
  }
}

__global__ void scan_kernel(const int* __restrict__ counts, int* __restrict__ offsets,
                            int* __restrict__ cursors) {
  if (threadIdx.x == 0 && blockIdx.x == 0) {
    int acc = 0;
    for (int e = 0; e < 16; ++e) { offsets[e] = acc; acc += counts[e]; cursors[e] = 0; }
    offsets[16] = acc;
  }
}

__global__ void scatter_kernel(const int* __restrict__ top_idx, const float* __restrict__ top_w,
                               const int* __restrict__ offsets, int* __restrict__ cursors,
                               int* __restrict__ token_ids, float* __restrict__ wts, int nAssign) {
  int a = blockIdx.x * blockDim.x + threadIdx.x;
  if (a >= nAssign) return;
  int e = top_idx[a];
  int pos = atomicAdd(&cursors[e], 1);
  int r = offsets[e] + pos;
  token_ids[r] = a >> 2;
  wts[r] = top_w[a];
}

// ---------------------------------------------------------------------------
// Gathered fused gate/up GEMM: A[r,f] = silu(X@Wg) * (X@Wu), bf16 out.
// 128x64 tile/block, 8 waves, each wave owns 2x2 16x16 tiles (8 WMMA/k-step),
// software-pipelined global->LDS staging. toks==null -> identity rows.
// W staging: each thread owns a k-pair of 4 columns -> packed ds_store_b32.
// ---------------------------------------------------------------------------
__global__ __launch_bounds__(256) void gateup_kernel(
    const unsigned short* __restrict__ xb, int ldx,
    const float* __restrict__ wg, const float* __restrict__ wu,
    long long wStrideE, int ldw, int kdim,
    unsigned short* __restrict__ Aout, int lda,
    const int* __restrict__ toks, const int* __restrict__ segOff,
    const int* __restrict__ counts, int fixedCount, int mBlocksPerE) {
  __shared__ __align__(16) unsigned short sX[BM * LDSROW];
  __shared__ __align__(16) unsigned short sG[BN * LDSROW];
  __shared__ __align__(16) unsigned short sU[BN * LDSROW];

  int e  = blockIdx.y / mBlocksPerE;
  int bm = blockIdx.y % mBlocksPerE;
  int cnt = counts ? counts[e] : fixedCount;
  int rowbase = bm * BM;
  if (rowbase >= cnt) return;
  int off = segOff ? segOff[e] : 0;
  int nbase = blockIdx.x * BN;
  const float* wgE = wg + (long long)e * wStrideE;
  const float* wuE = wu + (long long)e * wStrideE;

  int tid = threadIdx.x, lane = tid & 31, w = tid >> 5;
  int half = lane >> 4, l15 = lane & 15;
  int mg = w >> 1;            // 0..3 -> m-tiles {2mg, 2mg+1} of 8
  int ng = w & 1;             // 0..1 -> n-tiles {2ng, 2ng+1} of 4

  v8f accG[2][2] = {};
  v8f accU[2][2] = {};

  // X staging: thread -> (row 0..127, 16-elem chunk)
  int xr = tid >> 1, xc = (tid & 1) * 16;
  int srcRow = rowbase + xr;
  int clampedRow = srcRow < cnt ? srcRow : (cnt - 1);
  int token = toks ? toks[off + clampedRow] : clampedRow;
  const unsigned short* xsrc = xb + (long long)token * ldx + xc;
  // W staging: thread -> (k-pair 2*kp, 4 cols), stored transposed [n][k] packed
  int wc = (tid & 15) * 4;          // column chunk (coalesced across half-wave)
  int k2 = (tid >> 4) * 2;          // even k of the pair
  const float* gsrc = wgE + (long long)k2 * ldw + nbase + wc;
  const float* usrc = wuE + (long long)k2 * ldw + nbase + wc;

  int nsteps = kdim / KT;
  uint4  px0 = *(const uint4*)(xsrc);
  uint4  px1 = *(const uint4*)(xsrc + 8);
  float4 pg0 = *(const float4*)(gsrc);
  float4 pg1 = *(const float4*)(gsrc + ldw);
  float4 pu0 = *(const float4*)(usrc);
  float4 pu1 = *(const float4*)(usrc + ldw);

  for (int s = 0; s < nsteps; ++s) {
    *(uint4*)(&sX[xr * LDSROW + xc])     = px0;
    *(uint4*)(&sX[xr * LDSROW + xc + 8]) = px1;
    float g0[4] = {pg0.x, pg0.y, pg0.z, pg0.w};
    float g1[4] = {pg1.x, pg1.y, pg1.z, pg1.w};
    float u0[4] = {pu0.x, pu0.y, pu0.z, pu0.w};
    float u1[4] = {pu1.x, pu1.y, pu1.z, pu1.w};
#pragma unroll
    for (int j = 0; j < 4; ++j) {
      *(unsigned int*)(&sG[(wc + j) * LDSROW + k2]) = pack_bf16(g0[j], g1[j]);
      *(unsigned int*)(&sU[(wc + j) * LDSROW + k2]) = pack_bf16(u0[j], u1[j]);
    }
    __syncthreads();

    if (s + 1 < nsteps) {               // prefetch next k-step during WMMAs
      long long k1 = (long long)(s + 1) * KT;
      px0 = *(const uint4*)(xsrc + k1);
      px1 = *(const uint4*)(xsrc + k1 + 8);
      const float* g2 = gsrc + k1 * ldw;
      const float* u2 = usrc + k1 * ldw;
      pg0 = *(const float4*)(g2); pg1 = *(const float4*)(g2 + ldw);
      pu0 = *(const float4*)(u2); pu1 = *(const float4*)(u2 + ldw);
    }

    FragBF a[2], bg[2], bu[2];
#pragma unroll
    for (int im = 0; im < 2; ++im) {    // A fragment: two 16B chunks per lane
      const unsigned short* arp = &sX[((mg * 2 + im) * 16 + l15) * LDSROW];
      a[im].q[0] = *(const uint4*)(arp + half * 16);
      a[im].q[1] = *(const uint4*)(arp + half * 16 + 8);
    }
#pragma unroll
    for (int in = 0; in < 2; ++in) {    // B fragment: contiguous 16 K per half
      const unsigned short* pG = &sG[((ng * 2 + in) * 16 + l15) * LDSROW + half * 16];
      bg[in].q[0] = *(const uint4*)(pG); bg[in].q[1] = *(const uint4*)(pG + 8);
      const unsigned short* pU = &sU[((ng * 2 + in) * 16 + l15) * LDSROW + half * 16];
      bu[in].q[0] = *(const uint4*)(pU); bu[in].q[1] = *(const uint4*)(pU + 8);
    }
#pragma unroll
    for (int im = 0; im < 2; ++im) {
#pragma unroll
      for (int in = 0; in < 2; ++in) {
        accG[im][in] = __builtin_amdgcn_wmma_f32_16x16x32_bf16(
            false, a[im].v, false, bg[in].v, (short)0, accG[im][in], false, false);
        accU[im][in] = __builtin_amdgcn_wmma_f32_16x16x32_bf16(
            false, a[im].v, false, bu[in].v, (short)0, accU[im][in], false, false);
      }
    }
    __syncthreads();
  }

#pragma unroll
  for (int im = 0; im < 2; ++im) {
#pragma unroll
    for (int v = 0; v < 8; ++v) {
      int r = rowbase + (mg * 2 + im) * 16 + v + half * 8;  // C/D: m = vgpr + 8*half
      if (r < cnt) {
        long long arow = (long long)(off + r) * lda;
#pragma unroll
        for (int in = 0; in < 2; ++in) {
          int col = nbase + (ng * 2 + in) * 16 + l15;
          Aout[arow + col] = f2bf(silu_mul(accG[im][in][v], accU[im][in][v]));
        }
      }
    }
  }
}

// ---------------------------------------------------------------------------
// Down projection: out[token,h] (+)= wt * (A @ Wd). atomicMode=1 -> atomicAdd
// with routing weight; atomicMode=0 -> plain store (shared expert path).
// ---------------------------------------------------------------------------
__global__ __launch_bounds__(256) void down_kernel(
    const unsigned short* __restrict__ A, int lda,
    const float* __restrict__ wd, long long wStrideE, int ldw, int kdim,
    float* __restrict__ out, int ldo,
    const int* __restrict__ toks, const int* __restrict__ segOff,
    const int* __restrict__ counts, const float* __restrict__ wts,
    int fixedCount, int mBlocksPerE, int atomicMode) {
  __shared__ __align__(16) unsigned short sX[BM * LDSROW];
  __shared__ __align__(16) unsigned short sW[BN * LDSROW];

  int e  = blockIdx.y / mBlocksPerE;
  int bm = blockIdx.y % mBlocksPerE;
  int cnt = counts ? counts[e] : fixedCount;
  int rowbase = bm * BM;
  if (rowbase >= cnt) return;
  int off = segOff ? segOff[e] : 0;
  int nbase = blockIdx.x * BN;
  const float* wdE = wd + (long long)e * wStrideE;

  int tid = threadIdx.x, lane = tid & 31, w = tid >> 5;
  int half = lane >> 4, l15 = lane & 15;
  int mg = w >> 1, ng = w & 1;

  v8f acc[2][2] = {};

  int xr = tid >> 1, xc = (tid & 1) * 16;
  int srcRow = rowbase + xr;
  int clampedRow = srcRow < cnt ? srcRow : (cnt - 1);
  const unsigned short* xsrc = A + (long long)(off + clampedRow) * lda + xc;
  int wc = (tid & 15) * 4;
  int k2 = (tid >> 4) * 2;
  const float* dsrc = wdE + (long long)k2 * ldw + nbase + wc;

  int nsteps = kdim / KT;
  uint4  px0 = *(const uint4*)(xsrc);
  uint4  px1 = *(const uint4*)(xsrc + 8);
  float4 pd0 = *(const float4*)(dsrc);
  float4 pd1 = *(const float4*)(dsrc + ldw);

  for (int s = 0; s < nsteps; ++s) {
    *(uint4*)(&sX[xr * LDSROW + xc])     = px0;
    *(uint4*)(&sX[xr * LDSROW + xc + 8]) = px1;
    float d0[4] = {pd0.x, pd0.y, pd0.z, pd0.w};
    float d1[4] = {pd1.x, pd1.y, pd1.z, pd1.w};
#pragma unroll
    for (int j = 0; j < 4; ++j)
      *(unsigned int*)(&sW[(wc + j) * LDSROW + k2]) = pack_bf16(d0[j], d1[j]);
    __syncthreads();

    if (s + 1 < nsteps) {
      long long k1 = (long long)(s + 1) * KT;
      px0 = *(const uint4*)(xsrc + k1);
      px1 = *(const uint4*)(xsrc + k1 + 8);
      const float* d2 = dsrc + k1 * ldw;
      pd0 = *(const float4*)(d2); pd1 = *(const float4*)(d2 + ldw);
    }

    FragBF a[2], b[2];
#pragma unroll
    for (int im = 0; im < 2; ++im) {
      const unsigned short* arp = &sX[((mg * 2 + im) * 16 + l15) * LDSROW];
      a[im].q[0] = *(const uint4*)(arp + half * 16);
      a[im].q[1] = *(const uint4*)(arp + half * 16 + 8);
    }
#pragma unroll
    for (int in = 0; in < 2; ++in) {
      const unsigned short* p = &sW[((ng * 2 + in) * 16 + l15) * LDSROW + half * 16];
      b[in].q[0] = *(const uint4*)(p); b[in].q[1] = *(const uint4*)(p + 8);
    }
#pragma unroll
    for (int im = 0; im < 2; ++im) {
#pragma unroll
      for (int in = 0; in < 2; ++in) {
        acc[im][in] = __builtin_amdgcn_wmma_f32_16x16x32_bf16(
            false, a[im].v, false, b[in].v, (short)0, acc[im][in], false, false);
      }
    }
    __syncthreads();
  }

#pragma unroll
  for (int im = 0; im < 2; ++im) {
#pragma unroll
    for (int v = 0; v < 8; ++v) {
      int r = rowbase + (mg * 2 + im) * 16 + v + half * 8;
      if (r < cnt) {
        if (atomicMode) {
          int token = toks[off + r];
          float wt  = wts[off + r];
#pragma unroll
          for (int in = 0; in < 2; ++in) {
            int col = nbase + (ng * 2 + in) * 16 + l15;
            atomicAdd(&out[(long long)token * ldo + col], acc[im][in][v] * wt);
          }
        } else {
#pragma unroll
          for (int in = 0; in < 2; ++in) {
            int col = nbase + (ng * 2 + in) * 16 + l15;
            out[(long long)r * ldo + col] = acc[im][in][v];
          }
        }
      }
    }
  }
}

// ---------------------------------------------------------------------------
extern "C" void kernel_launch(void* const* d_in, const int* in_sizes, int n_in,
                              void* d_out, int out_size, void* d_ws, size_t ws_size,
                              hipStream_t stream) {
  (void)in_sizes; (void)n_in; (void)out_size; (void)ws_size;
  const float* x   = (const float*)d_in[0];   // [T,H]
  const float* gw  = (const float*)d_in[1];   // [E,H]
  const float* wgt = (const float*)d_in[2];   // [E,H,F]
  const float* wup = (const float*)d_in[3];   // [E,H,F]
  const float* wdn = (const float*)d_in[4];   // [E,F,H]
  const float* wsg = (const float*)d_in[5];   // [H,Fs]
  const float* wsu = (const float*)d_in[6];   // [H,Fs]
  const float* wsd = (const float*)d_in[7];   // [Fs,H]
  float* out = (float*)d_out;

  const int T = 2048, H = 2048, E = 16, F = 1408, Fs = 2816, K = 4;

  char* wsp = (char*)d_ws;
  size_t o = 0;
  auto carve = [&](size_t bytes) {
    void* p = wsp + o;
    o = (o + bytes + 255) & ~(size_t)255;
    return p;
  };
  int*   counts    = (int*)carve(16 * 4);
  int*   offsets   = (int*)carve(17 * 4);
  int*   cursors   = (int*)carve(16 * 4);
  int*   top_idx   = (int*)carve((size_t)T * K * 4);
  float* top_w     = (float*)carve((size_t)T * K * 4);
  int*   token_ids = (int*)carve((size_t)T * K * 4);
  float* wts       = (float*)carve((size_t)T * K * 4);
  unsigned short* xb = (unsigned short*)carve((size_t)T * H * 2);        // 8 MB
  unsigned short* Ar = (unsigned short*)carve((size_t)T * K * F * 2);    // 22 MiB
  unsigned short* As = (unsigned short*)carve((size_t)T * Fs * 2);       // 11.5 MB

  hipMemsetAsync(counts, 0, 16 * 4, stream);
  cvt_kernel<<<(T * H / 4 + 255) / 256, 256, 0, stream>>>(x, xb, (long long)T * H);
  gate_kernel<<<T / 8, 256, 0, stream>>>(x, gw, H, T, top_idx, top_w, counts);
  scan_kernel<<<1, 64, 0, stream>>>(counts, offsets, cursors);
  scatter_kernel<<<(T * K + 255) / 256, 256, 0, stream>>>(top_idx, top_w, offsets, cursors,
                                                          token_ids, wts, T * K);

  const int MBE = T / BM;   // m-blocks per expert (worst case all tokens -> one expert)

  // routed gate/up (gathered per-expert segments; blocks past count early-exit)
  dim3 g1(F / BN, E * MBE);
  gateup_kernel<<<g1, 256, 0, stream>>>(xb, H, wgt, wup, (long long)H * F, F, H,
                                        Ar, F, token_ids, offsets, counts, 0, MBE);
  // shared expert gate/up (identity rows)
  dim3 g2(Fs / BN, MBE);
  gateup_kernel<<<g2, 256, 0, stream>>>(xb, H, wsg, wsu, 0, Fs, H,
                                        As, Fs, nullptr, nullptr, nullptr, T, MBE);
  // shared down first: plain store fully writes out[T,H]
  dim3 g3(H / BN, MBE);
  down_kernel<<<g3, 256, 0, stream>>>(As, Fs, wsd, 0, H, Fs, out, H,
                                      nullptr, nullptr, nullptr, nullptr, T, MBE, 0);
  // routed down: weighted atomic accumulate
  dim3 g4(H / BN, E * MBE);
  down_kernel<<<g4, 256, 0, stream>>>(Ar, F, wdn, (long long)F * H, H, F, out, H,
                                      token_ids, offsets, counts, wts, 0, MBE, 1);
}
